// SparseHilbertSSM_83631603188091
// MI455X (gfx1250) — matrix operand order repaired and checked by reading, not compiled, not run on hardware
//
#include <hip/hip_runtime.h>

#define NDIRS 6
#define NLAYS 3
#define BB    4
#define LLEN  2048
#define ROWS  (BB*LLEN)      // 8192
#define HIDD  256
#define DINN  512
#define NHH   8
#define HDIM  64
#define DSTT  64
#define CONVD 640
#define DPROJ 1160
#define DPAD  1216           // padded to multiple of 64 (block N tile)
#define FDIM  1536
#define NCLS  11

typedef __attribute__((ext_vector_type(16))) _Float16 v16h;
typedef __attribute__((ext_vector_type(8)))  float    v8f;

#if defined(__AMDGCN__) && __has_builtin(__builtin_amdgcn_tensor_load_to_lds)
#define HAVE_TDM 1
#endif

#ifdef HAVE_TDM
typedef __attribute__((ext_vector_type(4))) unsigned int u32x4_t;
typedef __attribute__((ext_vector_type(8))) int i32x8_t;
typedef __attribute__((ext_vector_type(4))) int i32x4_t;

// Issue one TDM load of a 32(rows) x 32(f16) tile (row stride = K elements)
// from global into LDS at byte offset ldsoff.  D# layout per CDNA5 ISA §8:
//  group0: count=1, lds_addr, global_addr[56:0], type=2
//  group1: data_size=1(2B), tensor_dim0=K, tensor_dim1=32, tile_dim0=32,
//          tile_dim1=32, tensor_dim0_stride=K, everything else 0
__device__ __forceinline__ void tdm_load_a_tile(const _Float16* gsrc, int K,
                                                unsigned ldsoff) {
  unsigned long long ga = (unsigned long long)(size_t)gsrc;
  u32x4_t g0;
  g0[0] = 1u;                                           // count=1, user mode
  g0[1] = ldsoff;                                       // lds_addr (bytes)
  g0[2] = (unsigned)ga;                                 // global_addr[31:0]
  g0[3] = (unsigned)((ga >> 32) & 0x01ffffffull) | (2u << 30); // [56:32] | type=2
  i32x8_t g1;
  g1[0] = (int)0x00010000u;                             // data_size=1 (2 bytes)
  g1[1] = (int)(((unsigned)K & 0xffffu) << 16);         // tensor_dim0[15:0]
  g1[2] = (int)((((unsigned)K >> 16) & 0xffffu) | (32u << 16)); // td0 hi | td1 lo
  g1[3] = (int)(32u << 16);                             // td1 hi=0 | tile_dim0=32
  g1[4] = (int)32u;                                     // tile_dim1=32, tile_dim2=0
  g1[5] = (int)(unsigned)K;                             // tensor_dim0_stride[31:0]
  g1[6] = 0;                                            // stride hi | stride1 lo
  g1[7] = 0;
  i32x4_t gz = {0, 0, 0, 0};
#if __clang_major__ >= 23
  i32x8_t gz8 = {0, 0, 0, 0, 0, 0, 0, 0};
  __builtin_amdgcn_tensor_load_to_lds(g0, g1, gz, gz, gz8, 0);
#else
  __builtin_amdgcn_tensor_load_to_lds(g0, g1, gz, gz, 0);
#endif
}
#endif // HAVE_TDM

// ---------------------------------------------------------------- utilities
__device__ __forceinline__ float blk_sum(float v, float* red) {
  int t = threadIdx.x;
  red[t] = v;
  __syncthreads();
  for (int s = blockDim.x >> 1; s > 0; s >>= 1) {
    if (t < s) red[t] += red[t + s];
    __syncthreads();
  }
  float r = red[0];
  __syncthreads();
  return r;
}

__device__ __forceinline__ float gelu_f(float v) {
  return 0.5f * v * (1.f + erff(v * 0.70710678118654752f));
}
__device__ __forceinline__ float silu_f(float v) {
  return v / (1.f + expf(-v));
}

// ------------------------------------------------- bitonic lexsort per (dir,b)
__global__ void sort_kernel(const int* __restrict__ coords,
                            int* __restrict__ idx, int* __restrict__ inv) {
  __shared__ unsigned int keys[LLEN];
  const int ord[NDIRS][3] = {{0,1,2},{0,2,1},{1,0,2},{1,2,0},{2,0,1},{2,1,0}};
  int db = blockIdx.x;            // d*BB + b
  int d = db / BB, b = db % BB;
  int a0 = ord[d][0], a1 = ord[d][1], a2 = ord[d][2];
  for (int l = threadIdx.x; l < LLEN; l += blockDim.x) {
    const int* c = coords + ((size_t)(b * LLEN + l)) * 4;
    unsigned k = ((unsigned)c[a0] << 25) | ((unsigned)c[a1] << 18) |
                 ((unsigned)c[a2] << 11) | (unsigned)l;
    keys[l] = k;
  }
  for (int ksz = 2; ksz <= LLEN; ksz <<= 1) {
    for (int j = ksz >> 1; j > 0; j >>= 1) {
      __syncthreads();
      for (int i = threadIdx.x; i < LLEN; i += blockDim.x) {
        int ixj = i ^ j;
        if (ixj > i) {
          unsigned ki = keys[i], kj = keys[ixj];
          bool up = ((i & ksz) == 0);
          if (up ? (ki > kj) : (ki < kj)) { keys[i] = kj; keys[ixj] = ki; }
        }
      }
    }
  }
  __syncthreads();
  for (int l = threadIdx.x; l < LLEN; l += blockDim.x) {
    int src = (int)(keys[l] & 2047u);
    idx[(size_t)db * LLEN + l]   = src;
    inv[(size_t)db * LLEN + src] = l;
  }
}

// ------------------------------------------------- input embed + LN + GELU
__global__ void embed_kernel(const float* __restrict__ re, const float* __restrict__ im,
                             const float* __restrict__ W, const float* __restrict__ bias,
                             const float* __restrict__ g, const float* __restrict__ be,
                             float* __restrict__ x) {
  __shared__ float red[256];
  int row = blockIdx.x;           // b*LLEN + l
  int c = threadIdx.x;            // 0..255
  const float* r = re + (size_t)row * 64;
  const float* q = im + (size_t)row * 64;
  float acc = bias[c];
  for (int k = 0; k < 64; ++k) acc += r[k] * W[k * HIDD + c];
  for (int k = 0; k < 64; ++k) acc += q[k] * W[(64 + k) * HIDD + c];
  float S  = blk_sum(acc, red);
  float S2 = blk_sum(acc * acc, red);
  float mean = S / HIDD;
  float var  = S2 / HIDD - mean * mean;
  float v = (acc - mean) * rsqrtf(var + 1e-5f) * g[c] + be[c];
  x[(size_t)row * HIDD + c] = gelu_f(v);
}

// ------------------------------------------------- gather / scatter
__global__ void gather_kernel(const float* __restrict__ x, const int* __restrict__ idx,
                              float* __restrict__ xd) {
  size_t t = (size_t)blockIdx.x * blockDim.x + threadIdx.x;
  if (t >= (size_t)NDIRS * ROWS * HIDD) return;
  int c = (int)(t & (HIDD - 1));
  size_t r = t >> 8;              // d*ROWS + b*LLEN + l
  int d = (int)(r / ROWS);
  int bl = (int)(r - (size_t)d * ROWS);
  int b = bl >> 11, l = bl & (LLEN - 1);
  int src = idx[(size_t)(d * BB + b) * LLEN + l];
  xd[t] = x[(((size_t)(b * LLEN + src)) << 8) + c];
}

__global__ void scatter_kernel(const float* __restrict__ xd, const int* __restrict__ inv,
                               float* __restrict__ md) {
  size_t t = (size_t)blockIdx.x * blockDim.x + threadIdx.x;
  if (t >= (size_t)ROWS * NDIRS * HIDD) return;
  int c = (int)(t & (HIDD - 1));
  size_t r = t >> 8;              // (b*LLEN+l)*NDIRS + d
  int d = (int)(r % NDIRS);
  size_t bl = r / NDIRS;
  int b = (int)(bl >> 11), l = (int)(bl & (LLEN - 1));
  int src = inv[(size_t)(d * BB + b) * LLEN + l];
  md[t] = xd[(((size_t)(d * ROWS + b * LLEN + src)) << 8) + c];
}

// ------------------------------------------------- generic row LN -> f16
__global__ void ln_f16_kernel(const float* __restrict__ in, const float* __restrict__ g,
                              const float* __restrict__ be, _Float16* __restrict__ out,
                              int rowlen) {
  __shared__ float red[256];
  int row = blockIdx.x;
  const float* p = in + (size_t)row * rowlen;
  float s = 0.f, s2 = 0.f;
  for (int c = threadIdx.x; c < rowlen; c += blockDim.x) {
    float v = p[c]; s += v; s2 += v * v;
  }
  float S  = blk_sum(s,  red);
  float S2 = blk_sum(s2, red);
  float mean = S / rowlen;
  float var  = S2 / rowlen - mean * mean;
  float inv = rsqrtf(var + 1e-5f);
  for (int c = threadIdx.x; c < rowlen; c += blockDim.x)
    out[(size_t)row * rowlen + c] = (_Float16)((p[c] - mean) * inv * g[c] + be[c]);
}

// ------------------------------------------------- pack weight to WMMA-B layout
// Bp[((kt*Nt+nt)*32+lane)*16 + j] = W[(kt*32 + j + (lane>=16?16:0)) * N + nt*16 + lane%16]
__global__ void pack_b_kernel(const float* __restrict__ W, _Float16* __restrict__ Bp,
                              int K, int N, int Npad) {
  int t = blockIdx.x * blockDim.x + threadIdx.x;
  int total = K * Npad;
  if (t >= total) return;
  int j    = t & 15;
  int lane = (t >> 4) & 31;
  int rest = t >> 9;              // kt*Nt + nt
  int Nt = Npad >> 4;
  int nt = rest % Nt;
  int kt = rest / Nt;
  int n = nt * 16 + (lane & 15);
  int k = kt * 32 + j + ((lane >> 4) << 4);
  float v = (n < N) ? W[(size_t)k * N + n] : 0.f;
  Bp[t] = (_Float16)v;
}

// ------------------------------------------------- WMMA f16 GEMM, C f32
// Block = 8 waves -> 32(M) x 64(N) macro tile.  A tile (32x32 f16) staged in
// LDS via Tensor Data Mover (double buffered), B read from packed layout.
// A: MxK row-major f16; Bp: packed (K/32, Npad/16, 32, 16) f16; C: MxN f32.
__global__ void gemm_wmma_f16(const _Float16* __restrict__ A,
                              const _Float16* __restrict__ Bp,
                              float* __restrict__ C,
                              int M, int K, int Npad, int N, int accumulate) {
  __shared__ _Float16 ldsA[2][32 * 32];
  const int lane = threadIdx.x & 31;
  const int wave = threadIdx.x >> 5;
  const int Nt = Npad >> 4;
  const int nBlkN = Npad >> 6;                 // N blocks of 64
  const int bm = blockIdx.x / nBlkN;           // 32-row block
  const int bn = blockIdx.x - bm * nBlkN;      // 64-col block
  const int mstrip = bm * 2 + (wave & 1);      // 16-row strip id
  const int nt = bn * 4 + (wave >> 1);         // 16-col tile id
  const int lm = lane & 15;
  const int hi = lane >> 4;                    // 0 or 1
  const int Kt = K >> 5;
  const int arow = (wave & 1) * 16 + lm;       // row inside 32-row LDS tile
  v8f acc = {0.f, 0.f, 0.f, 0.f, 0.f, 0.f, 0.f, 0.f};
  const _Float16* Abase = A + (size_t)bm * 32 * K;
#ifdef HAVE_TDM
  if (threadIdx.x < 32) {
    tdm_load_a_tile(Abase, K, (unsigned)(size_t)&ldsA[0][0]);
    __builtin_amdgcn_s_wait_tensorcnt(0);
  }
  __syncthreads();
#endif
  for (int kt = 0; kt < Kt; ++kt) {
    union { v16h v; float4 f[2]; } a, b;
#ifdef HAVE_TDM
    const int cur = kt & 1;
    if (threadIdx.x < 32 && kt + 1 < Kt)
      tdm_load_a_tile(Abase + (kt + 1) * 32, K,
                      (unsigned)(size_t)&ldsA[cur ^ 1][0]);
    const _Float16* la = &ldsA[cur][arow * 32 + hi * 16];
    a.f[0] = *(const float4*)(la);
    a.f[1] = *(const float4*)(la + 8);
#else
    const _Float16* pa = Abase + (size_t)arow * K + kt * 32 + hi * 16;
    a.f[0] = *(const float4*)(pa);
    a.f[1] = *(const float4*)(pa + 8);
#endif
    const _Float16* pb = Bp + (((size_t)kt * Nt + nt) * 32 + lane) * 16;
    b.f[0] = *(const float4*)(pb);
    b.f[1] = *(const float4*)(pb + 8);
    acc = __builtin_amdgcn_wmma_f32_16x16x32_f16(false, a.v, false, b.v,
                                                 (short)0, acc, false, false);
#ifdef HAVE_TDM
    if (threadIdx.x < 32 && kt + 1 < Kt)
      __builtin_amdgcn_s_wait_tensorcnt(0);
    __syncthreads();
#endif
  }
  int n = nt * 16 + lm;
  if (n < N) {
    int mbase = mstrip * 16 + hi * 8;
    #pragma unroll
    for (int r = 0; r < 8; ++r) {
      float* cp = C + (size_t)(mbase + r) * N + n;
      if (accumulate) *cp += acc[r]; else *cp = acc[r];
    }
  }
}

// ------------------------------------------------- conv(4) + SiLU + dt/dA
__global__ void conv_silu_dt_kernel(const float* __restrict__ zx,
                                    const float* __restrict__ convW,
                                    const float* __restrict__ convb,
                                    const float* __restrict__ dtb,
                                    const float* __restrict__ Alog,
                                    float* __restrict__ xBCc,
                                    float* __restrict__ dt, float* __restrict__ dA) {
  int t = blockIdx.x * blockDim.x + threadIdx.x;
  const int total = ROWS * (CONVD + NHH);
  if (t >= total) return;
  int row = t / (CONVD + NHH);
  int c   = t - row * (CONVD + NHH);
  int b = row >> 11, l = row & (LLEN - 1);
  if (c < CONVD) {
    float acc = convb[c];
    #pragma unroll
    for (int k = 0; k < 4; ++k) {
      int ls = l + k - 3;
      if (ls >= 0)
        acc += convW[c * 4 + k] * zx[((size_t)(b * LLEN + ls)) * DPROJ + DINN + c];
    }
    xBCc[(size_t)row * CONVD + c] = silu_f(acc);
  } else {
    int h = c - CONVD;
    float draw = zx[(size_t)row * DPROJ + DINN + CONVD + h] + dtb[h];
    float dtv = (draw > 20.f) ? draw : log1pf(expf(draw));
    dt[(size_t)row * NHH + h] = dtv;
    dA[(size_t)row * NHH + h] = expf(-expf(Alog[h]) * dtv);
  }
}

// ------------------------------------------------- selective scan
// block = (b,h); thread t: p = t>>2, n-chunk g = t&3 (16 states each)
__global__ void scan_kernel(const float* __restrict__ xBCc,
                            const float* __restrict__ dt, const float* __restrict__ dA,
                            float* __restrict__ y) {
  int bh = blockIdx.x;            // 0..BB*NHH-1
  int b = bh >> 3, h = bh & 7;
  int t = threadIdx.x;            // 0..255
  int p = t >> 2, g = t & 3;
  float hst[16];
  #pragma unroll
  for (int i = 0; i < 16; ++i) hst[i] = 0.f;
  for (int l = 0; l < LLEN; ++l) {
    size_t row = (size_t)(b * LLEN + l);
    const float* xr = xBCc + row * CONVD;
    float dtv = dt[row * NHH + h];
    float dAv = dA[row * NHH + h];
    float xv  = xr[h * HDIM + p];
    const float* Bv = xr + DINN + g * 16;
    const float* Cv = xr + DINN + DSTT + g * 16;
    float part = 0.f;
    float dx = dtv * xv;
    #pragma unroll
    for (int i = 0; i < 16; ++i) {
      hst[i] = hst[i] * dAv + dx * Bv[i];
      part += hst[i] * Cv[i];
    }
    part += __shfl_xor(part, 1, 32);
    part += __shfl_xor(part, 2, 32);
    if (g == 0) y[row * DINN + h * HDIM + p] = part;
  }
}

// ------------------------------------------------- D*xh + gate + RMSNorm -> f16
__global__ void gate_rms_kernel(const float* __restrict__ y, const float* __restrict__ xBCc,
                                const float* __restrict__ zx, const float* __restrict__ D,
                                const float* __restrict__ gnw, _Float16* __restrict__ yh) {
  __shared__ float red[256];
  __shared__ float vals[DINN];
  int row = blockIdx.x;
  float s2 = 0.f;
  for (int c = threadIdx.x; c < DINN; c += blockDim.x) {
    float v = y[(size_t)row * DINN + c] + D[c >> 6] * xBCc[(size_t)row * CONVD + c];
    float z = zx[(size_t)row * DPROJ + c];
    v *= silu_f(z);
    vals[c] = v;
    s2 += v * v;
  }
  float S2 = blk_sum(s2, red);
  float r = rsqrtf(S2 / DINN + 1e-5f);
  for (int c = threadIdx.x; c < DINN; c += blockDim.x)
    yh[(size_t)row * DINN + c] = (_Float16)(vals[c] * r * gnw[c]);
}

// ------------------------------------------------- bias + gelu (in place)
__global__ void bias_gelu_kernel(float* __restrict__ a, const float* __restrict__ bias,
                                 int n, int rowlen) {
  int t = blockIdx.x * blockDim.x + threadIdx.x;
  if (t >= n) return;
  float v = a[t] + bias[t % rowlen];
  a[t] = gelu_f(v);
}

// ------------------------------------------------- global mean over L
__global__ void mean_kernel(const float* __restrict__ fused, float* __restrict__ pm) {
  int t = blockIdx.x * blockDim.x + threadIdx.x;   // b*HIDD + c
  if (t >= BB * HIDD) return;
  int b = t >> 8, c = t & (HIDD - 1);
  float s = 0.f;
  for (int l = 0; l < LLEN; ++l) s += fused[((size_t)(b * LLEN + l)) * HIDD + c];
  pm[t] = s * (1.f / (float)LLEN);
}

// ------------------------------------------------- classifier head (B=4 rows)
__global__ void cls_kernel(const float* __restrict__ pm,
                           const float* __restrict__ lng, const float* __restrict__ lnb,
                           const float* __restrict__ W1, const float* __restrict__ b1,
                           const float* __restrict__ W2, const float* __restrict__ b2,
                           const float* __restrict__ W3, const float* __restrict__ b3,
                           float* __restrict__ out) {
  __shared__ float p[3 * HIDD];
  __shared__ float h1[2 * HIDD];
  __shared__ float h2[HIDD];
  __shared__ float red[256];
  int b = blockIdx.x;
  for (int c = threadIdx.x; c < 3 * HIDD; c += blockDim.x)
    p[c] = pm[b * HIDD + (c & (HIDD - 1))];
  __syncthreads();
  float s = 0.f, s2 = 0.f;
  for (int c = threadIdx.x; c < 3 * HIDD; c += blockDim.x) {
    float v = p[c]; s += v; s2 += v * v;
  }
  float S  = blk_sum(s,  red);
  float S2 = blk_sum(s2, red);
  float mean = S / (3 * HIDD);
  float var  = S2 / (3 * HIDD) - mean * mean;
  float inv = rsqrtf(var + 1e-5f);
  __syncthreads();
  for (int c = threadIdx.x; c < 3 * HIDD; c += blockDim.x)
    p[c] = (p[c] - mean) * inv * lng[c] + lnb[c];
  __syncthreads();
  for (int o = threadIdx.x; o < 2 * HIDD; o += blockDim.x) {
    float a = b1[o];
    for (int k = 0; k < 3 * HIDD; ++k) a += p[k] * W1[k * (2 * HIDD) + o];
    h1[o] = gelu_f(a);
  }
  __syncthreads();
  for (int o = threadIdx.x; o < HIDD; o += blockDim.x) {
    float a = b2[o];
    for (int k = 0; k < 2 * HIDD; ++k) a += h1[k] * W2[k * HIDD + o];
    h2[o] = gelu_f(a);
  }
  __syncthreads();
  if (threadIdx.x < NCLS) {
    int o = threadIdx.x;
    float a = b3[o];
    for (int k = 0; k < HIDD; ++k) a += h2[k] * W3[k * NCLS + o];
    out[b * NCLS + o] = a;
  }
}

// ================================================================ launcher
extern "C" void kernel_launch(void* const* d_in, const int* in_sizes, int n_in,
                              void* d_out, int out_size, void* d_ws, size_t ws_size,
                              hipStream_t stream) {
  const float* vec_real = (const float*)d_in[0];
  const float* vec_imag = (const float*)d_in[1];
  const int*   coords   = (const int*)d_in[2];
  const float* inp_W    = (const float*)d_in[3];
  const float* inp_b    = (const float*)d_in[4];
  const float* inp_ln_g = (const float*)d_in[5];
  const float* inp_ln_b = (const float*)d_in[6];
  const float* blk_ln_g = (const float*)d_in[7];
  const float* blk_ln_b = (const float*)d_in[8];
  const float* blk_inW  = (const float*)d_in[9];
  const float* blk_convW= (const float*)d_in[10];
  const float* blk_convb= (const float*)d_in[11];
  const float* blk_dtb  = (const float*)d_in[12];
  const float* blk_Alog = (const float*)d_in[13];
  const float* blk_D    = (const float*)d_in[14];
  const float* blk_gnw  = (const float*)d_in[15];
  const float* blk_outW = (const float*)d_in[16];
  const float* fus_ln_g = (const float*)d_in[17];
  const float* fus_ln_b = (const float*)d_in[18];
  const float* fus_W    = (const float*)d_in[19];
  const float* fus_b    = (const float*)d_in[20];
  const float* cls_ln_g = (const float*)d_in[21];
  const float* cls_ln_b = (const float*)d_in[22];
  const float* cls_W1   = (const float*)d_in[23];
  const float* cls_b1   = (const float*)d_in[24];
  const float* cls_W2   = (const float*)d_in[25];
  const float* cls_b2   = (const float*)d_in[26];
  const float* cls_W3   = (const float*)d_in[27];
  const float* cls_b3   = (const float*)d_in[28];
  float* out = (float*)d_out;

  // ---------------- workspace carve-up (256B aligned)
  char* ws = (char*)d_ws;
  size_t off = 0;
  auto carve = [&](size_t bytes) -> char* {
    char* p = ws + off;
    off += (bytes + 255) & ~(size_t)255;
    return p;
  };
  int*   idx     = (int*)  carve((size_t)NDIRS * BB * LLEN * 4);
  int*   inv     = (int*)  carve((size_t)NDIRS * BB * LLEN * 4);
  float* x       = (float*)carve((size_t)ROWS * HIDD * 4);
  float* xd      = (float*)carve((size_t)NDIRS * ROWS * HIDD * 4);
  _Float16* inWp = (_Float16*)carve((size_t)NDIRS * NLAYS * HIDD * DPAD * 2);
  _Float16* outWp= (_Float16*)carve((size_t)NDIRS * NLAYS * DINN * HIDD * 2);
  _Float16* fusWp= (_Float16*)carve((size_t)FDIM * HIDD * 2);
  float* pm      = (float*)carve((size_t)BB * HIDD * 4);
  // overlapped "big" region: layer-loop temporaries vs post-loop buffers
  size_t szA = 0;
  size_t o_xn   = szA; szA += ((size_t)ROWS * HIDD * 2 + 255) & ~(size_t)255;
  size_t o_zx   = szA; szA += ((size_t)ROWS * DPROJ * 4 + 255) & ~(size_t)255;
  size_t o_xbc  = szA; szA += ((size_t)ROWS * CONVD * 4 + 255) & ~(size_t)255;
  size_t o_dt   = szA; szA += ((size_t)ROWS * NHH * 4 + 255) & ~(size_t)255;
  size_t o_dA   = szA; szA += ((size_t)ROWS * NHH * 4 + 255) & ~(size_t)255;
  size_t o_y    = szA; szA += ((size_t)ROWS * DINN * 4 + 255) & ~(size_t)255;
  size_t o_yh   = szA; szA += ((size_t)ROWS * DINN * 2 + 255) & ~(size_t)255;
  size_t szB = 0;
  size_t o_md   = szB; szB += ((size_t)ROWS * FDIM * 4 + 255) & ~(size_t)255;
  size_t o_mdh  = szB; szB += ((size_t)ROWS * FDIM * 2 + 255) & ~(size_t)255;
  size_t o_fz   = szB; szB += ((size_t)ROWS * HIDD * 4 + 255) & ~(size_t)255;
  char* big = carve(szA > szB ? szA : szB);
  _Float16* xn   = (_Float16*)(big + o_xn);
  float*    zx   = (float*)   (big + o_zx);
  float*    xBCc = (float*)   (big + o_xbc);
  float*    dtbuf= (float*)   (big + o_dt);
  float*    dAbuf= (float*)   (big + o_dA);
  float*    ybuf = (float*)   (big + o_y);
  _Float16* yh   = (_Float16*)(big + o_yh);
  float*    md   = (float*)   (big + o_md);
  _Float16* mdh  = (_Float16*)(big + o_mdh);
  float*    fz   = (float*)   (big + o_fz);
  (void)ws_size; (void)in_sizes; (void)n_in; (void)out_size;

  // ---------------- 1) lexsort permutations
  sort_kernel<<<NDIRS * BB, 1024, 0, stream>>>(coords, idx, inv);

  // ---------------- 2) input embed
  embed_kernel<<<ROWS, 256, 0, stream>>>(vec_real, vec_imag, inp_W, inp_b,
                                         inp_ln_g, inp_ln_b, x);

  // ---------------- 3) gather per direction
  {
    size_t tot = (size_t)NDIRS * ROWS * HIDD;
    gather_kernel<<<(int)((tot + 255) / 256), 256, 0, stream>>>(x, idx, xd);
  }

  // ---------------- 4) pack weights to f16 WMMA-B layout
  for (int dl = 0; dl < NDIRS * NLAYS; ++dl) {
    pack_b_kernel<<<(HIDD * DPAD + 255) / 256, 256, 0, stream>>>(
        blk_inW + (size_t)dl * HIDD * DPROJ, inWp + (size_t)dl * HIDD * DPAD,
        HIDD, DPROJ, DPAD);
    pack_b_kernel<<<(DINN * HIDD + 255) / 256, 256, 0, stream>>>(
        blk_outW + (size_t)dl * DINN * HIDD, outWp + (size_t)dl * DINN * HIDD,
        DINN, HIDD, HIDD);
  }
  pack_b_kernel<<<(FDIM * HIDD + 255) / 256, 256, 0, stream>>>(fus_W, fusWp,
                                                               FDIM, HIDD, HIDD);

  // ---------------- 5) mamba blocks: layers outer, directions inner
  for (int layer = 0; layer < NLAYS; ++layer) {
    for (int d = 0; d < NDIRS; ++d) {
      int dl = d * NLAYS + layer;
      float* xdd = xd + (size_t)d * ROWS * HIDD;
      // LN -> f16
      ln_f16_kernel<<<ROWS, 256, 0, stream>>>(xdd, blk_ln_g + (size_t)dl * HIDD,
                                              blk_ln_b + (size_t)dl * HIDD, xn, HIDD);
      // in-proj GEMM: 8192 x 256 x 1160 (pad 1216)
      {
        int blocks = (ROWS / 32) * (DPAD / 64);
        gemm_wmma_f16<<<blocks, 256, 0, stream>>>(
            xn, inWp + (size_t)dl * HIDD * DPAD, zx, ROWS, HIDD, DPAD, DPROJ, 0);
      }
      // conv + silu + dt/dA
      {
        int tot = ROWS * (CONVD + NHH);
        conv_silu_dt_kernel<<<(tot + 255) / 256, 256, 0, stream>>>(
            zx, blk_convW + (size_t)dl * CONVD * 4, blk_convb + (size_t)dl * CONVD,
            blk_dtb + (size_t)dl * NHH, blk_Alog + (size_t)dl * NHH,
            xBCc, dtbuf, dAbuf);
      }
      // selective scan
      scan_kernel<<<BB * NHH, 256, 0, stream>>>(xBCc, dtbuf, dAbuf, ybuf);
      // D*xh + gate + RMS -> f16
      gate_rms_kernel<<<ROWS, 256, 0, stream>>>(ybuf, xBCc, zx,
                                                blk_D + (size_t)dl * NHH,
                                                blk_gnw + (size_t)dl * DINN, yh);
      // out-proj GEMM, accumulate into residual xd[d]
      {
        int blocks = (ROWS / 32) * (HIDD / 64);
        gemm_wmma_f16<<<blocks, 256, 0, stream>>>(
            yh, outWp + (size_t)dl * DINN * HIDD, xdd, ROWS, DINN, HIDD, HIDD, 1);
      }
    }
  }

  // ---------------- 6) inverse permute + concat
  {
    size_t tot = (size_t)ROWS * NDIRS * HIDD;
    scatter_kernel<<<(int)((tot + 255) / 256), 256, 0, stream>>>(xd, inv, md);
  }

  // ---------------- 7) fusion LN + GEMM + bias-gelu
  ln_f16_kernel<<<ROWS, 256, 0, stream>>>(md, fus_ln_g, fus_ln_b, mdh, FDIM);
  {
    int blocks = (ROWS / 32) * (HIDD / 64);
    gemm_wmma_f16<<<blocks, 256, 0, stream>>>(mdh, fusWp, fz,
                                              ROWS, FDIM, HIDD, HIDD, 0);
  }
  bias_gelu_kernel<<<(ROWS * HIDD + 255) / 256, 256, 0, stream>>>(fz, fus_b,
                                                                  ROWS * HIDD, HIDD);

  // ---------------- 8) pooling (all scales == global mean) + classifier
  mean_kernel<<<(BB * HIDD + 255) / 256, 256, 0, stream>>>(fz, pm);
  cls_kernel<<<BB, 256, 0, stream>>>(pm, cls_ln_g, cls_ln_b, cls_W1, cls_b1,
                                     cls_W2, cls_b2, cls_W3, cls_b3, out);
}
